// CSPNetFull_25280177504325
// MI455X (gfx1250) — matrix operand order repaired and checked by reading, not compiled, not run on hardware
//
#include <hip/hip_runtime.h>
#include <hip/hip_bf16.h>
#include <cstdint>

typedef __attribute__((ext_vector_type(16))) _Float16 v16h;
typedef __attribute__((ext_vector_type(8)))  _Float16 v8h;
typedef __attribute__((ext_vector_type(8)))  float    v8f;

// ---------------- problem constants ----------------
#define CN_H     512
#define CN_TD    256
#define CN_NF    128
#define CN_L     6
#define CN_B     256
#define CN_NA    20
#define CN_N     (CN_B * CN_NA)         // 5120
#define CN_E     (CN_B * CN_NA * CN_NA) // 102400
#define CN_EIN   1801                   // 2H + 9 + NF*2*3
#define CN_TAIL  800                    // 9 lat + 768 fourier + 23 pad  (25 k-tiles)
#define CN_KPAD  1824                   // 512 + 512 + 800

// ================= small utility kernels =================

// f32 weights -> f16 pre-swizzled into WMMA B-fragment tile order:
// dst[((kt*(Ntot/16) + ntile)*32 + lane)*16 + el] = W[k,n]
//   kk = (lane>>4)*8 + (el<8 ? el : 8+el) ; k = kt*32+kk ; n = ntile*16 + (lane&15)
__global__ __launch_bounds__(256) void cvtswz_kernel(const float* __restrict__ src,
                                                     _Float16* __restrict__ dst,
                                                     int Ktiles, int Kreal, int Ntot) {
  const int total = Ktiles * Ntot * 32;
  for (int idx = blockIdx.x * 256 + threadIdx.x; idx < total; idx += gridDim.x * 256) {
    const int kt    = idx / (Ntot * 32);
    const int rem   = idx % (Ntot * 32);
    const int ntile = rem / 512;
    const int lane  = (rem % 512) / 16;
    const int el    = rem % 16;
    const int kk    = ((lane >> 4) << 3) + (el < 8 ? el : 8 + el);
    const int k     = kt * 32 + kk;
    const int n     = ntile * 16 + (lane & 15);
    dst[idx] = (k < Kreal) ? (_Float16)src[(size_t)k * Ntot + n] : (_Float16)0.f;
  }
}

// sinusoidal time embedding [B,256]
__global__ __launch_bounds__(256) void temb_kernel(const float* __restrict__ t,
                                                   float* __restrict__ temb) {
  const int g = blockIdx.x, c = threadIdx.x;
  const float scale = 9.210340371976184f / 127.0f;   // log(1e4)/(half-1)
  const int i = c & 127;
  const float a = t[g] * __expf(-scale * (float)i);
  temb[g * CN_TD + c] = (c < 128) ? sinf(a) : cosf(a);
}

// lattice inner product [B,9]
__global__ void latip_kernel(const float* __restrict__ lat, float* __restrict__ latip) {
  const int g = blockIdx.x, t = threadIdx.x;
  if (t < 9) {
    const int i = t / 3, j = t % 3;
    float s = 0.f;
    for (int k = 0; k < 3; ++k) s += lat[g*9 + i*3 + k] * lat[g*9 + j*3 + k];
    latip[g*9 + t] = s;
  }
}

// edge tail segment, f16 [E,800] = [lat_ip(9) | sin(384) | cos(384) | 0(23)]
__global__ __launch_bounds__(256) void tail_kernel(const float* __restrict__ frac,
                                                   const int* __restrict__ srcI,
                                                   const int* __restrict__ dstI,
                                                   const float* __restrict__ latip,
                                                   _Float16* __restrict__ tail) {
  const int e = blockIdx.x;
  __shared__ float fd[3];
  if (threadIdx.x < 3) {
    const int d = threadIdx.x;
    const float x = frac[dstI[e]*3 + d] - frac[srcI[e]*3 + d];
    fd[d] = x - floorf(x);              // mod 1
  }
  __syncthreads();
  const int g9 = (e / (CN_NA * CN_NA)) * 9;
  for (int r = 0; r < 4; ++r) {
    const int q = threadIdx.x + 256 * r;          // 0..1023
    if (q >= CN_TAIL) break;
    float val;
    if (q < 9) {
      val = latip[g9 + q];
    } else if (q < 777) {
      const int idx = q - 9;
      const int sc  = idx / 384;                  // 0=sin 1=cos
      const int rem = idx % 384;
      const int d   = rem / CN_NF, f = rem % CN_NF;
      const float arg = fd[d] * 6.283185307179586f * (float)f;
      val = sc ? cosf(arg) : sinf(arg);
    } else {
      val = 0.f;                                  // pad to 800
    }
    tail[(size_t)e * CN_TAIL + q] = (_Float16)val;
  }
}

// pack [emb_table[atom-1], t_emb[node2graph]] -> f16 [N,768]
__global__ __launch_bounds__(256) void a0_kernel(const float* __restrict__ emb,
                                                 const float* __restrict__ temb,
                                                 const int* __restrict__ atom,
                                                 _Float16* __restrict__ A0) {
  const int n = blockIdx.x;
  const int c = blockIdx.y * 256 + threadIdx.x;   // 0..767
  float v;
  if (c < CN_H) {
    int a = atom[n] - 1; a = a < 0 ? 0 : (a > 99 ? 99 : a);
    v = emb[a * CN_H + c];
  } else {
    v = temb[(n / CN_NA) * CN_TD + (c - CN_H)];
  }
  A0[(size_t)n * 768 + c] = (_Float16)v;
}

// LayerNorm, wave-per-row (wave32); optional f16 and/or f32 outputs
__global__ __launch_bounds__(256) void ln_kernel(const float* __restrict__ x,
                                                 const float* __restrict__ g,
                                                 const float* __restrict__ b,
                                                 _Float16* __restrict__ out16,
                                                 float* __restrict__ out32) {
  const int lane = threadIdx.x & 31, wave = threadIdx.x >> 5;
  const int row = blockIdx.x * 8 + wave;
  const float* xr = x + (size_t)row * CN_H;
  float v[16], s = 0.f, s2 = 0.f;
  for (int i = 0; i < 16; ++i) {
    const float t = xr[lane + 32 * i];
    v[i] = t; s += t; s2 += t * t;
  }
  for (int m = 16; m >= 1; m >>= 1) {          // wave32 butterfly reduce
    s  += __shfl_xor(s,  m, 32);
    s2 += __shfl_xor(s2, m, 32);
  }
  const float mean = s * (1.f / CN_H);
  const float var  = s2 * (1.f / CN_H) - mean * mean;
  const float inv  = rsqrtf(var + 1e-5f);
  for (int i = 0; i < 16; ++i) {
    const int c = lane + 32 * i;
    const float y = (v[i] - mean) * inv * g[c] + b[c];
    if (out16) out16[(size_t)row * CN_H + c] = (_Float16)y;
    if (out32) out32[(size_t)row * CN_H + c] = y;
  }
}

// scatter-mean over 20 consecutive edges per src node (block structure)
__global__ __launch_bounds__(256) void agg_kernel(const _Float16* __restrict__ e2,
                                                  _Float16* __restrict__ agg) {
  const int n = blockIdx.x;
  const int c = blockIdx.y * 256 + threadIdx.x;
  float s = 0.f;
  for (int j = 0; j < CN_NA; ++j)
    s += (float)e2[(size_t)(n * CN_NA + j) * CN_H + c];
  agg[(size_t)n * CN_H + c] = (_Float16)(s * (1.f / CN_NA));
}

__global__ __launch_bounds__(256) void pos_kernel(const float* __restrict__ hF,
                                                  const float* __restrict__ Wc,
                                                  float* __restrict__ out) {
  const int idx = blockIdx.x * 256 + threadIdx.x;     // N*3 = 15360
  const int n = idx / 3, c = idx % 3;
  float s = 0.f;
  for (int k = 0; k < CN_H; ++k) s += hF[(size_t)n * CN_H + k] * Wc[k * 3 + c];
  out[idx] = s;
}

__global__ __launch_bounds__(256) void ghmean_kernel(const float* __restrict__ hF,
                                                     float* __restrict__ gh) {
  const int g = blockIdx.x;
  const int c = blockIdx.y * 256 + threadIdx.x;
  float s = 0.f;
  for (int j = 0; j < CN_NA; ++j) s += hF[(size_t)(g * CN_NA + j) * CN_H + c];
  gh[(size_t)g * CN_H + c] = s * (1.f / CN_NA);
}

__global__ __launch_bounds__(256) void cell_kernel(const float* __restrict__ gh,
                                                   const float* __restrict__ Wl,
                                                   const float* __restrict__ lat,
                                                   float* __restrict__ out) {
  const int idx = blockIdx.x * 256 + threadIdx.x;     // B*9 = 2304
  if (idx >= CN_B * 9) return;
  const int g = idx / 9, i = (idx % 9) / 3, k = idx % 3;
  const float* ghg = gh + (size_t)g * CN_H;
  float r = 0.f;
  for (int j = 0; j < 3; ++j) {
    float t = 0.f;
    for (int q = 0; q < CN_H; ++q) t += ghg[q] * Wl[q * 9 + i * 3 + j];
    r += t * lat[g * 9 + j * 3 + k];
  }
  out[idx] = r;
}

// ================= WMMA GEMM (LDS-free, direct fragment loads) =================
// 256 threads = 8 wave32 waves. Block tile: M=128 (wave = m-tile), N=32
// (2 accumulators per wave). A fragment = two contiguous 16B chunks of the
// lane's row: k in [k0+8*half,+8) and [k0+16+8*half,+8). Segment boundaries
// (512/1024) are k-tile aligned, so no bounds checks, no LDS, no barriers.
// B is pre-swizzled into fragment-tile order: one contiguous v16h per lane.
//
// AMODE: 0 = plain f16 A [M,lda]
//        1 = edge features: [hn[src](16kt) | hn[dst](16kt) | tail(25kt)]
//        2 = node features: [hn(16kt) | agg(16kt)]
// EPI:   0 = +bias, f32 store
//        1 = +bias, SiLU, f16 store
//        2 = +bias, SiLU, f32 accumulate (residual)
template<int AMODE, int EPI>
__global__ __launch_bounds__(256) void wmma_gemm_kernel(
    const _Float16* __restrict__ A, int lda, int Ktiles,     // AMODE 0 only
    const _Float16* __restrict__ Bsw,                        // swizzled weights
    const float* __restrict__ bias,
    float* __restrict__ Cf, _Float16* __restrict__ Ch, int Ntot,
    const _Float16* __restrict__ hn, const int* __restrict__ srcI,
    const int* __restrict__ dstI, const _Float16* __restrict__ tail,
    const _Float16* __restrict__ aggp)
{
  const int lane = threadIdx.x & 31;
  const int mt   = threadIdx.x >> 5;                 // wave id = m-tile 0..7
  const int rr   = lane & 15;
  const int half = lane >> 4;
  const int rowA = blockIdx.x * 128 + mt * 16 + rr;  // lane's A row
  const int aoff = 8 * half;                         // chunk offset within k-tile

  // B: fragment (kt, ntile) at ((kt*(Ntot/16) + ntile)*512); lane stride 16
  const size_t bstep = (size_t)Ntot * 32;
  const _Float16* bp = Bsw + ((size_t)blockIdx.y * 2) * 512 + lane * 16;

  v8f acc0 = {}, acc1 = {};

  auto runseg = [&](const _Float16* rowp, int kts, bool pf) {
    for (int t = 0; t < kts; ++t) {
      const _Float16* p = rowp + t * 32 + aoff;
      union { v16h v; struct { v8h lo; v8h hi; } s; } a;
      a.s.lo = *(const v8h*)(p);
      a.s.hi = *(const v8h*)(p + 16);
      if (pf) __builtin_prefetch(p + 512, 0, 1);     // stream tail ahead (HBM)
      const v16h b0 = *(const v16h*)(bp);
      const v16h b1 = *(const v16h*)(bp + 512);
      acc0 = __builtin_amdgcn_wmma_f32_16x16x32_f16(false, a.v, false, b0,
                                                    (short)0, acc0, false, false);
      acc1 = __builtin_amdgcn_wmma_f32_16x16x32_f16(false, a.v, false, b1,
                                                    (short)0, acc1, false, false);
      bp += bstep;
    }
  };

  if (AMODE == 0) {
    runseg(A + (size_t)rowA * lda, Ktiles, false);
  } else if (AMODE == 1) {
    runseg(hn + (size_t)srcI[rowA] * CN_H, CN_H / 32, false);
    runseg(hn + (size_t)dstI[rowA] * CN_H, CN_H / 32, false);
    runseg(tail + (size_t)rowA * CN_TAIL, CN_TAIL / 32, true);
  } else {
    runseg(hn + (size_t)rowA * CN_H, CN_H / 32, false);
    runseg(aggp + (size_t)rowA * CN_H, CN_H / 32, false);
  }

  // epilogue: C layout VGPR r -> M = r + 8*(lane>=16), N = lane%16
  const int n  = lane & 15;
  const int mh = lane >> 4;
  const int row0 = blockIdx.x * 128 + mt * 16;
  for (int nt = 0; nt < 2; ++nt) {
    const v8f& acc = nt ? acc1 : acc0;
    const int col  = blockIdx.y * 32 + nt * 16 + n;
    const float bi = bias[col];
    for (int r8 = 0; r8 < 8; ++r8) {
      const int row = row0 + r8 + 8 * mh;
      const float v = acc[r8] + bi;
      if (EPI == 0) {
        Cf[(size_t)row * Ntot + col] = v;
      } else {
        const float sv = v / (1.f + __expf(-v));     // SiLU
        if (EPI == 1) Ch[(size_t)row * Ntot + col] = (_Float16)sv;
        else          Cf[(size_t)row * Ntot + col] += sv;   // residual
      }
    }
  }
}

// ================= host orchestration =================
extern "C" void kernel_launch(void* const* d_in, const int* in_sizes, int n_in,
                              void* d_out, int out_size, void* d_ws, size_t ws_size,
                              hipStream_t stream) {
  const float* t          = (const float*)d_in[0];
  const float* frac       = (const float*)d_in[1];
  const float* lattices   = (const float*)d_in[2];
  const float* emb_table  = (const float*)d_in[3];
  const float* W_latent   = (const float*)d_in[4];
  const float* b_latent   = (const float*)d_in[5];
  const float* ln_g       = (const float*)d_in[6];
  const float* ln_b       = (const float*)d_in[7];
  const float* edge_W1    = (const float*)d_in[8];
  const float* edge_b1    = (const float*)d_in[9];
  const float* edge_W2    = (const float*)d_in[10];
  const float* edge_b2    = (const float*)d_in[11];
  const float* node_W1    = (const float*)d_in[12];
  const float* node_b1    = (const float*)d_in[13];
  const float* node_W2    = (const float*)d_in[14];
  const float* node_b2    = (const float*)d_in[15];
  const float* fln_g      = (const float*)d_in[16];
  const float* fln_b      = (const float*)d_in[17];
  const float* W_coord    = (const float*)d_in[18];
  const float* W_lat      = (const float*)d_in[19];
  const int*   atom_types = (const int*)d_in[20];
  const int*   edge_index = (const int*)d_in[21];
  const int* srcI = edge_index;
  const int* dstI = edge_index + CN_E;

  // -------- workspace carve-out --------
  char* wsb = (char*)d_ws;
  size_t off = 0;
  auto carve = [&](size_t bytes) -> void* {
    void* p = wsb + off;
    off = (off + bytes + 255) & ~(size_t)255;
    return p;
  };
  float*    h      = (float*)   carve((size_t)CN_N * CN_H * 4);
  _Float16* hn16   = (_Float16*)carve((size_t)CN_N * CN_H * 2);
  float*    hF     = (float*)   carve((size_t)CN_N * CN_H * 4);
  _Float16* tail16 = (_Float16*)carve((size_t)CN_E * CN_TAIL * 2);
  float*    latip  = (float*)   carve((size_t)CN_B * 9 * 4);
  float*    temb   = (float*)   carve((size_t)CN_B * CN_TD * 4);
  _Float16* A016   = (_Float16*)carve((size_t)CN_N * 768 * 2);
  _Float16* e116   = (_Float16*)carve((size_t)CN_E * CN_H * 2);
  _Float16* e216   = (_Float16*)carve((size_t)CN_E * CN_H * 2);
  _Float16* agg16  = (_Float16*)carve((size_t)CN_N * CN_H * 2);
  _Float16* n116   = (_Float16*)carve((size_t)CN_N * CN_H * 2);
  float*    graphh = (float*)   carve((size_t)CN_B * CN_H * 4);
  _Float16* Wlatsw = (_Float16*)carve((size_t)768 * CN_H * 2);
  _Float16* eW1sw  = (_Float16*)carve((size_t)CN_L * CN_KPAD * CN_H * 2);
  _Float16* eW2sw  = (_Float16*)carve((size_t)CN_L * CN_H * CN_H * 2);
  _Float16* nW1sw  = (_Float16*)carve((size_t)CN_L * 2 * CN_H * CN_H * 2);
  _Float16* nW2sw  = (_Float16*)carve((size_t)CN_L * CN_H * CN_H * 2);

  const _Float16* nh16 = nullptr; const int* ni = nullptr;

  // -------- weight conversion: f32 -> f16, pre-swizzled to B-fragment order --
  cvtswz_kernel<<<1024, 256, 0, stream>>>(W_latent, Wlatsw, 768/32, 768, CN_H);
  for (int l = 0; l < CN_L; ++l) {
    cvtswz_kernel<<<1024, 256, 0, stream>>>(edge_W1 + (size_t)l*CN_EIN*CN_H,
        eW1sw + (size_t)l*CN_KPAD*CN_H, CN_KPAD/32, CN_EIN, CN_H);
    cvtswz_kernel<<<1024, 256, 0, stream>>>(edge_W2 + (size_t)l*CN_H*CN_H,
        eW2sw + (size_t)l*CN_H*CN_H, CN_H/32, CN_H, CN_H);
    cvtswz_kernel<<<1024, 256, 0, stream>>>(node_W1 + (size_t)l*2*CN_H*CN_H,
        nW1sw + (size_t)l*2*CN_H*CN_H, 2*CN_H/32, 2*CN_H, CN_H);
    cvtswz_kernel<<<1024, 256, 0, stream>>>(node_W2 + (size_t)l*CN_H*CN_H,
        nW2sw + (size_t)l*CN_H*CN_H, CN_H/32, CN_H, CN_H);
  }

  // -------- hoisted precompute --------
  temb_kernel<<<CN_B, CN_TD, 0, stream>>>(t, temb);
  latip_kernel<<<CN_B, 16, 0, stream>>>(lattices, latip);
  tail_kernel<<<CN_E, 256, 0, stream>>>(frac, srcI, dstI, latip, tail16);
  a0_kernel<<<dim3(CN_N, 3), 256, 0, stream>>>(emb_table, temb, atom_types, A016);

  // h0 = [emb, t_emb] @ W_latent + b   (no activation)
  wmma_gemm_kernel<0, 0><<<dim3(CN_N / 128, CN_H / 32), 256, 0, stream>>>(
      A016, 768, 768 / 32, Wlatsw, b_latent, h, (_Float16*)nullptr, CN_H,
      nh16, ni, ni, nh16, nh16);

  // -------- message-passing layers --------
  for (int l = 0; l < CN_L; ++l) {
    ln_kernel<<<CN_N / 8, 256, 0, stream>>>(h, ln_g + l * CN_H, ln_b + l * CN_H,
                                            hn16, (float*)nullptr);
    // edge MLP layer 1: fused gather [E,1824]x[1824,512]
    wmma_gemm_kernel<1, 1><<<dim3(CN_E / 128, CN_H / 32), 256, 0, stream>>>(
        (const _Float16*)nullptr, 0, 0, eW1sw + (size_t)l * CN_KPAD * CN_H,
        edge_b1 + l * CN_H, (float*)nullptr, e116, CN_H,
        hn16, srcI, dstI, tail16, nh16);
    // edge MLP layer 2: [E,512]x[512,512]
    wmma_gemm_kernel<0, 1><<<dim3(CN_E / 128, CN_H / 32), 256, 0, stream>>>(
        e116, CN_H, CN_H / 32, eW2sw + (size_t)l * CN_H * CN_H,
        edge_b2 + l * CN_H, (float*)nullptr, e216, CN_H,
        nh16, ni, ni, nh16, nh16);
    // scatter-mean (src-major runs of 20 edges)
    agg_kernel<<<dim3(CN_N, 2), 256, 0, stream>>>(e216, agg16);
    // node MLP layer 1: gather [hn|agg], [N,1024]x[1024,512]
    wmma_gemm_kernel<2, 1><<<dim3(CN_N / 128, CN_H / 32), 256, 0, stream>>>(
        (const _Float16*)nullptr, 0, 0, nW1sw + (size_t)l * 2 * CN_H * CN_H,
        node_b1 + l * CN_H, (float*)nullptr, n116, CN_H,
        hn16, ni, ni, nh16, agg16);
    // node MLP layer 2 + residual: h += silu([N,512]x[512,512]+b)
    wmma_gemm_kernel<0, 2><<<dim3(CN_N / 128, CN_H / 32), 256, 0, stream>>>(
        n116, CN_H, CN_H / 32, nW2sw + (size_t)l * CN_H * CN_H,
        node_b2 + l * CN_H, h, (_Float16*)nullptr, CN_H,
        nh16, ni, ni, nh16, nh16);
  }

  // -------- readout --------
  ln_kernel<<<CN_N / 8, 256, 0, stream>>>(h, fln_g, fln_b, (_Float16*)nullptr, hF);
  float* out = (float*)d_out;
  pos_kernel<<<(CN_N * 3) / 256, 256, 0, stream>>>(hF, W_coord, out);
  ghmean_kernel<<<dim3(CN_B, 2), 256, 0, stream>>>(hF, graphh);
  cell_kernel<<<(CN_B * 9 + 255) / 256, 256, 0, stream>>>(graphh, W_lat, lattices,
                                                          out + CN_N * 3);
  (void)in_sizes; (void)n_in; (void)out_size; (void)ws_size;
}
